// MetricPoseLoss_11690900980317
// MI455X (gfx1250) — compile-verified, matrix operand order checked by reading, not run on hardware
//
#include <hip/hip_runtime.h>
#include <hip/hip_bf16.h>
#include <math.h>

typedef __attribute__((ext_vector_type(16))) _Float16 v16h;
typedef __attribute__((ext_vector_type(8)))  float    v8f;
typedef __attribute__((ext_vector_type(4)))  float    f32x4;

#define B_          32
#define NK_         1024
#define NSAMP       512
#define IT_MATCHES_ 4
#define IT_RANSAC_  8
#define NUM_CORR_   5
#define INLIER_TH   0.15f
#define SCORE_TEMP_ 10.0f
#define TH_OUT_     0.35f
#define MAXLOSSNULL 0.5f
#define SOFTCLIP_M  0.5f
#define BETA_       (5.0f / 0.15f)

// ---------------- RNG helpers (deterministic hash) ----------------
__device__ __forceinline__ uint32_t hash3(uint32_t a, uint32_t b, uint32_t c) {
  uint32_t h = a * 0x9E3779B1u + b * 0x85EBCA77u + c * 0xC2B2AE3Du + 0x27220A95u;
  h ^= h >> 16; h *= 0x7FEB352Du;
  h ^= h >> 15; h *= 0x846CA68Bu;
  h ^= h >> 16;
  return h;
}
__device__ __forceinline__ float uniform01(uint32_t h) {
  return (float)(h >> 8) * (1.0f / 16777216.0f) + 1e-7f;
}
__device__ __forceinline__ float sigmoidf(float x) {
  return 1.0f / (1.0f + __expf(-x));
}

// ============================================================================
// Kernel 1: fused streaming gumbel-sampling.  One pass over the 128MB matches
// tensor computing all IT_MATCHES iterations at once (exp-race argmin).
// One wave per (b, slot); slot covers 2048 contiguous elements of the 1M row.
// ============================================================================
__global__ __launch_bounds__(256) void sample_kernel(
    const float* __restrict__ matches, int* __restrict__ sidx_ws,
    float* __restrict__ w_ws) {
  const int wave = threadIdx.x >> 5;
  const int lane = threadIdx.x & 31;
  const int slot = blockIdx.x * 8 + wave;   // 0 .. B*512-1
  const int b = slot >> 9;
  const int s = slot & 511;
  const f32x4* mp = reinterpret_cast<const f32x4*>(
      matches + (size_t)b * (NK_ * NK_) + (size_t)s * 2048);

  float bk[IT_MATCHES_]; int bi[IT_MATCHES_]; float bv[IT_MATCHES_];
#pragma unroll
  for (int t = 0; t < IT_MATCHES_; ++t) { bk[t] = 3.4e38f; bi[t] = 0; bv[t] = 1.0f; }

  for (int chunk = 0; chunk < 16; ++chunk) {
    f32x4 v = __builtin_nontemporal_load(mp + chunk * 32 + lane);
    const int e0 = s * 2048 + chunk * 128 + lane * 4;
    float mv[4] = {v.x, v.y, v.z, v.w};
#pragma unroll
    for (int j = 0; j < 4; ++j) {
      const int idx = e0 + j;
      const float inv_m = 1.0f / (mv[j] + 1e-12f);
#pragma unroll
      for (int t = 0; t < IT_MATCHES_; ++t) {
        float u = uniform01(hash3((uint32_t)idx,
                                  (uint32_t)(t * 0x51ED270Bu + 0x2545F491u),
                                  (uint32_t)b));
        float key = (-__logf(u)) * inv_m;   // exp-race == gumbel argmax of log m
        if (key < bk[t]) { bk[t] = key; bi[t] = idx; bv[t] = mv[j]; }
      }
    }
  }
  // wave32 argmin reduction
#pragma unroll
  for (int t = 0; t < IT_MATCHES_; ++t) {
#pragma unroll
    for (int off = 16; off > 0; off >>= 1) {
      float ok = __shfl_down(bk[t], off, 32);
      int   oi = __shfl_down(bi[t], off, 32);
      float ov = __shfl_down(bv[t], off, 32);
      if (ok < bk[t]) { bk[t] = ok; bi[t] = oi; bv[t] = ov; }
    }
  }
  if (lane == 0) {
#pragma unroll
    for (int t = 0; t < IT_MATCHES_; ++t) {
      int o = ((t * B_ + b) << 9) + s;
      sidx_ws[o] = bi[t];
      w_ws[o] = bv[t];
    }
  }
}

// ============================================================================
// Kernel 2: backprojection with analytic K^-1 (zero-skew pinhole).
// ============================================================================
__global__ __launch_bounds__(256) void backproject_kernel(
    const int* __restrict__ sidx_ws,
    const float* __restrict__ kps0, const float* __restrict__ depth0,
    const float* __restrict__ kps1, const float* __restrict__ depth1,
    const float* __restrict__ K0, const float* __restrict__ K1,
    float* __restrict__ Xws, float* __restrict__ Yws) {
  int gid = blockIdx.x * 256 + threadIdx.x;     // < IT_MATCHES*B*512
  int rem = gid & (B_ * NSAMP - 1);
  int b = rem >> 9;
  int sidx = sidx_ws[gid];
  int i0 = sidx >> 10;
  int i1 = sidx & (NK_ - 1);

  float fx0 = K0[b * 9 + 0], cx0 = K0[b * 9 + 2];
  float fy0 = K0[b * 9 + 4], cy0 = K0[b * 9 + 5];
  float u0 = kps0[b * 2 * NK_ + i0];
  float v0 = kps0[b * 2 * NK_ + NK_ + i0];
  float d0 = depth0[b * 2 * NK_ + i0];
  Xws[gid * 3 + 0] = d0 * (u0 - cx0) / fx0;
  Xws[gid * 3 + 1] = d0 * (v0 - cy0) / fy0;
  Xws[gid * 3 + 2] = d0;

  float fx1 = K1[b * 9 + 0], cx1 = K1[b * 9 + 2];
  float fy1 = K1[b * 9 + 4], cy1 = K1[b * 9 + 5];
  float u1 = kps1[b * 2 * NK_ + i1];
  float v1 = kps1[b * 2 * NK_ + NK_ + i1];
  float d1 = depth1[b * 2 * NK_ + i1];
  Yws[gid * 3 + 0] = d1 * (u1 - cx1) / fx1;
  Yws[gid * 3 + 1] = d1 * (v1 - cy1) / fy1;
  Yws[gid * 3 + 2] = d1;
}

// ---------------- 3x3 SVD -> Procrustes rotation (single thread) ------------
__device__ void svd3_rotation(const float* Hin, float* Rout) {
  float H[3][3];
  for (int i = 0; i < 3; ++i)
    for (int j = 0; j < 3; ++j) H[i][j] = Hin[i * 3 + j];
  float S[3][3];
  for (int i = 0; i < 3; ++i)
    for (int j = 0; j < 3; ++j) {
      float a = 0.0f;
      for (int k = 0; k < 3; ++k) a += H[k][i] * H[k][j];
      S[i][j] = a;
    }
  float V[3][3] = {{1, 0, 0}, {0, 1, 0}, {0, 0, 1}};
  for (int sweep = 0; sweep < 8; ++sweep) {
    for (int pq = 0; pq < 3; ++pq) {
      int p = (pq == 2) ? 1 : 0;
      int q = (pq == 0) ? 1 : 2;
      float apq = S[p][q];
      if (fabsf(apq) > 1e-14f) {
        float tau = (S[q][q] - S[p][p]) / (2.0f * apq);
        float tt = ((tau >= 0.0f) ? 1.0f : -1.0f) /
                   (fabsf(tau) + sqrtf(1.0f + tau * tau));
        float cc = 1.0f / sqrtf(1.0f + tt * tt);
        float ss = tt * cc;
        for (int k = 0; k < 3; ++k) {
          float skp = S[k][p], skq = S[k][q];
          S[k][p] = cc * skp - ss * skq;
          S[k][q] = ss * skp + cc * skq;
        }
        for (int k = 0; k < 3; ++k) {
          float spk = S[p][k], sqk = S[q][k];
          S[p][k] = cc * spk - ss * sqk;
          S[q][k] = ss * spk + cc * sqk;
        }
        for (int k = 0; k < 3; ++k) {
          float vkp = V[k][p], vkq = V[k][q];
          V[k][p] = cc * vkp - ss * vkq;
          V[k][q] = ss * vkp + cc * vkq;
        }
      }
    }
  }
  float eig[3] = {S[0][0], S[1][1], S[2][2]};
  for (int i = 0; i < 2; ++i)
    for (int j = i + 1; j < 3; ++j)
      if (eig[j] > eig[i]) {
        float te = eig[i]; eig[i] = eig[j]; eig[j] = te;
        for (int k = 0; k < 3; ++k) {
          float tv = V[k][i]; V[k][i] = V[k][j]; V[k][j] = tv;
        }
      }
  float U[3][3];
  for (int i = 0; i < 3; ++i) {
    float hv[3];
    for (int r = 0; r < 3; ++r)
      hv[r] = H[r][0] * V[0][i] + H[r][1] * V[1][i] + H[r][2] * V[2][i];
    float sg = sqrtf(fmaxf(eig[i], 0.0f));
    float inv = (sg > 1e-8f) ? (1.0f / sg) : 0.0f;
    for (int r = 0; r < 3; ++r) U[r][i] = hv[r] * inv;
  }
  {  // re-orthogonalize last left-singular column (handles rank deficiency)
    float cx = U[1][0] * U[2][1] - U[2][0] * U[1][1];
    float cy = U[2][0] * U[0][1] - U[0][0] * U[2][1];
    float cz = U[0][0] * U[1][1] - U[1][0] * U[0][1];
    float nn = sqrtf(cx * cx + cy * cy + cz * cz);
    if (nn > 1e-12f) { U[0][2] = cx / nn; U[1][2] = cy / nn; U[2][2] = cz / nn; }
  }
  float M[3][3];
  for (int i = 0; i < 3; ++i)
    for (int j = 0; j < 3; ++j)
      M[i][j] = V[i][0] * U[j][0] + V[i][1] * U[j][1] + V[i][2] * U[j][2];
  float d = M[0][0] * (M[1][1] * M[2][2] - M[1][2] * M[2][1]) -
            M[0][1] * (M[1][0] * M[2][2] - M[1][2] * M[2][0]) +
            M[0][2] * (M[1][0] * M[2][1] - M[1][1] * M[2][0]);
  for (int i = 0; i < 3; ++i) V[i][2] *= d;
  for (int i = 0; i < 3; ++i)
    for (int j = 0; j < 3; ++j)
      Rout[i * 3 + j] = V[i][0] * U[j][0] + V[i][1] * U[j][1] + V[i][2] * U[j][2];
}

// ---------------- block reductions (wave32 shuffle + LDS) -------------------
template <int N>
__device__ void waveBlockReduce(float* v, float* sred, int tid) {
  int lane = tid & 31, wave = tid >> 5;
#pragma unroll
  for (int off = 16; off > 0; off >>= 1) {
#pragma unroll
    for (int j = 0; j < N; ++j) v[j] += __shfl_down(v[j], off, 32);
  }
  if (lane == 0) {
#pragma unroll
    for (int j = 0; j < N; ++j) sred[wave * 12 + j] = v[j];
  }
  __syncthreads();
  if (tid == 0) {
#pragma unroll
    for (int j = 0; j < N; ++j) {
      float a = 0.0f;
      for (int w2 = 0; w2 < 8; ++w2) a += sred[w2 * 12 + j];
      v[j] = a;
    }
  }
  __syncthreads();
}

__device__ void blockArgMin(float v, int idx, float* sf, int* si, int tid) {
  int lane = tid & 31, wave = tid >> 5;
#pragma unroll
  for (int off = 16; off > 0; off >>= 1) {
    float ov = __shfl_down(v, off, 32);
    int oi = __shfl_down(idx, off, 32);
    if (ov < v) { v = ov; idx = oi; }
  }
  if (lane == 0) { sf[wave] = v; si[wave] = idx; }
  __syncthreads();
  if (tid == 0) {
    float bv2 = sf[0]; int bi2 = si[0];
    for (int w2 = 1; w2 < 8; ++w2)
      if (sf[w2] < bv2) { bv2 = sf[w2]; bi2 = si[w2]; }
    sf[8] = bv2; si[8] = bi2;
  }
  __syncthreads();
}

// ---------------- WMMA distance field: || R*X + t - Y || over 512 pts -------
// A (16x32 f16): 16 points as rows [x y z 1 0...]; B (32x16 f16): [R^T; t].
// Each wave owns 4 tiles of 16 points -> v_wmma_f32_16x16x32_f16 per tile.
__device__ void compute_dists(const float* shX, const float* shY,
                              const float* Rm, const float* tv,
                              float* shDist, float* shP, int tid) {
  const int wave = tid >> 5;
  const int lane = tid & 31;
  const int N = lane & 15;
  const bool hi = lane >= 16;

  v16h Bm = {};
  if (!hi && N < 3) {
    Bm[0] = (_Float16)Rm[N * 3 + 0];
    Bm[1] = (_Float16)Rm[N * 3 + 1];
    Bm[2] = (_Float16)Rm[N * 3 + 2];
    Bm[3] = (_Float16)tv[N];
  }
  for (int tb = 0; tb < 4; ++tb) {
    const int m0 = (wave * 4 + tb) * 16;
    v16h A = {};
    if (!hi) {
      const int M = m0 + lane;
      A[0] = (_Float16)shX[M * 3 + 0];
      A[1] = (_Float16)shX[M * 3 + 1];
      A[2] = (_Float16)shX[M * 3 + 2];
      A[3] = (_Float16)1.0f;
    }
    v8f c = {};
    c = __builtin_amdgcn_wmma_f32_16x16x32_f16(false, A, false, Bm, (short)0, c,
                                               false, false);
    if (N < 3) {
#pragma unroll
      for (int r = 0; r < 8; ++r) {
        int m = (hi ? 8 : 0) + r;
        shP[wave * 64 + m * 4 + N] = c[r];
      }
    }
    __syncthreads();
    if (!hi) {
      const int M = m0 + lane;
      float dx = shP[wave * 64 + lane * 4 + 0] - shY[M * 3 + 0];
      float dy = shP[wave * 64 + lane * 4 + 1] - shY[M * 3 + 1];
      float dz = shP[wave * 64 + lane * 4 + 2] - shY[M * 3 + 2];
      shDist[M] = sqrtf(dx * dx + dy * dy + dz * dz);
    }
    __syncthreads();
  }
}

// ---------------- weighted Procrustes over 512 LDS-resident points ----------
__device__ void procrustes_block(const float* shX, const float* shY,
                                 const float* inl, float* sred, float* shMu,
                                 float* outRT, int tid) {
  float a[7];
  for (int j = 0; j < 7; ++j) a[j] = 0.0f;
  for (int i = tid; i < NSAMP; i += 256) {
    float wv = inl[i];
    a[0] += wv;
    a[1] += wv * shX[i * 3 + 0]; a[2] += wv * shX[i * 3 + 1]; a[3] += wv * shX[i * 3 + 2];
    a[4] += wv * shY[i * 3 + 0]; a[5] += wv * shY[i * 3 + 1]; a[6] += wv * shY[i * 3 + 2];
  }
  waveBlockReduce<7>(a, sred, tid);
  if (tid == 0) {
    float sw = a[0] + 1e-8f;
    float inv = 1.0f / sw;
    shMu[0] = a[1] * inv; shMu[1] = a[2] * inv; shMu[2] = a[3] * inv;
    shMu[3] = a[4] * inv; shMu[4] = a[5] * inv; shMu[5] = a[6] * inv;
    shMu[6] = sw;
  }
  __syncthreads();
  float mx0 = shMu[0], mx1 = shMu[1], mx2 = shMu[2];
  float my0 = shMu[3], my1 = shMu[4], my2 = shMu[5];
  float invsw = 1.0f / shMu[6];
  float h[9];
  for (int j = 0; j < 9; ++j) h[j] = 0.0f;
  for (int i = tid; i < NSAMP; i += 256) {
    float wn = inl[i] * invsw;
    float x0 = shX[i * 3 + 0] - mx0, x1 = shX[i * 3 + 1] - mx1, x2 = shX[i * 3 + 2] - mx2;
    float y0 = shY[i * 3 + 0] - my0, y1 = shY[i * 3 + 1] - my1, y2 = shY[i * 3 + 2] - my2;
    h[0] += wn * x0 * y0; h[1] += wn * x0 * y1; h[2] += wn * x0 * y2;
    h[3] += wn * x1 * y0; h[4] += wn * x1 * y1; h[5] += wn * x1 * y2;
    h[6] += wn * x2 * y0; h[7] += wn * x2 * y1; h[8] += wn * x2 * y2;
  }
  waveBlockReduce<9>(h, sred, tid);
  if (tid == 0) {
    float R[9];
    svd3_rotation(h, R);
    for (int j = 0; j < 9; ++j) outRT[j] = R[j];
    outRT[9]  = my0 - (R[0] * mx0 + R[1] * mx1 + R[2] * mx2);
    outRT[10] = my1 - (R[3] * mx0 + R[4] * mx1 + R[5] * mx2);
    outRT[11] = my2 - (R[6] * mx0 + R[7] * mx1 + R[8] * mx2);
  }
  __syncthreads();
}

// ============================================================================
// Kernel 3: per-batch RANSAC + refinement + pose loss.  One block per batch,
// X/Y/w LDS-resident, distances via v_wmma_f32_16x16x32_f16.
// ============================================================================
__global__ __launch_bounds__(256) void ransac_kernel(
    const float* __restrict__ Xws, const float* __restrict__ Yws,
    const float* __restrict__ w_ws, const float* __restrict__ T,
    float* __restrict__ out) {
  __shared__ float shX[NSAMP * 3], shY[NSAMP * 3];
  __shared__ float shW[NSAMP], shKey[NSAMP], shDist[NSAMP], shInl[NSAMP];
  __shared__ float shP[8 * 64];
  __shared__ float sred[112];
  __shared__ float sargf[9];
  __shared__ int sargi[9];
  __shared__ float shRd[12], shRn[12];
  __shared__ float shMu[8];
  __shared__ float shScores[IT_RANSAC_], shLv[IT_RANSAC_];
  __shared__ int shRidx[NUM_CORR_];

  const int b = blockIdx.x;
  const int tid = threadIdx.x;
  float total = 0.0f;  // meaningful on tid 0

  for (int it = 0; it < IT_MATCHES_; ++it) {
    for (int i = tid; i < NSAMP; i += 256) {
      int g = ((it * B_ + b) << 9) + i;
      shW[i] = w_ws[g];
      shX[i * 3 + 0] = Xws[g * 3 + 0];
      shX[i * 3 + 1] = Xws[g * 3 + 1];
      shX[i * 3 + 2] = Xws[g * 3 + 2];
      shY[i * 3 + 0] = Yws[g * 3 + 0];
      shY[i * 3 + 1] = Yws[g * 3 + 1];
      shY[i * 3 + 2] = Yws[g * 3 + 2];
    }
    __syncthreads();

    for (int kk = 0; kk < IT_RANSAC_; ++kk) {
      // gumbel top-NUM_CORR over w (exp-race keys), plus reset inlier mask
      for (int i = tid; i < NSAMP; i += 256) {
        float u = uniform01(hash3((uint32_t)(i + it * 0x9E37u),
                                  (uint32_t)(kk * 0x85EBu + 0x1234u),
                                  (uint32_t)b));
        shKey[i] = (-__logf(u)) / (shW[i] + 1e-12f);
        shInl[i] = 0.0f;
      }
      __syncthreads();
      for (int p = 0; p < NUM_CORR_; ++p) {
        float lv = 3.4e38f; int li = 0;
        for (int i = tid; i < NSAMP; i += 256)
          if (shKey[i] < lv) { lv = shKey[i]; li = i; }
        blockArgMin(lv, li, sargf, sargi, tid);
        if (tid == 0) {
          int bi2 = sargi[8];
          shRidx[p] = bi2;
          shKey[bi2] = 3.4e38f;
          shInl[bi2] = 1.0f;
        }
        __syncthreads();
      }
      // minimal 5-point Procrustes (unit weights) -> R_pre, t_pre  (tid 0)
      if (tid == 0) {
        float mx[3] = {0, 0, 0}, my[3] = {0, 0, 0};
        for (int p = 0; p < NUM_CORR_; ++p) {
          int i = shRidx[p];
          for (int k = 0; k < 3; ++k) {
            mx[k] += 0.2f * shX[i * 3 + k];
            my[k] += 0.2f * shY[i * 3 + k];
          }
        }
        float h[9] = {0, 0, 0, 0, 0, 0, 0, 0, 0};
        for (int p = 0; p < NUM_CORR_; ++p) {
          int i = shRidx[p];
          float xc[3], yc[3];
          for (int k = 0; k < 3; ++k) {
            xc[k] = shX[i * 3 + k] - mx[k];
            yc[k] = shY[i * 3 + k] - my[k];
          }
          for (int r = 0; r < 3; ++r)
            for (int c2 = 0; c2 < 3; ++c2) h[r * 3 + c2] += 0.2f * xc[r] * yc[c2];
        }
        float R5[9];
        svd3_rotation(h, R5);
        for (int j = 0; j < 9; ++j) shRd[j] = R5[j];
        shRd[9]  = my[0] - (R5[0] * mx[0] + R5[1] * mx[1] + R5[2] * mx[2]);
        shRd[10] = my[1] - (R5[3] * mx[0] + R5[4] * mx[1] + R5[5] * mx[2]);
        shRd[11] = my[2] - (R5[6] * mx[0] + R5[7] * mx[1] + R5[8] * mx[2]);
        sred[101] = 0.0f;  // inl_pre
      }
      __syncthreads();

      // hypothesis score via WMMA distance field
      compute_dists(shX, shY, shRd, shRd + 9, shDist, shP, tid);
      {
        float sacc[1] = {0.0f};
        for (int i = tid; i < NSAMP; i += 256)
          sacc[0] += sigmoidf(BETA_ * (INLIER_TH - shDist[i]));
        waveBlockReduce<1>(sacc, sred, tid);
        if (tid == 0) shScores[kk] = sacc[0];
        __syncthreads();
      }

      // refinement iterations
      for (int r = 0; r < 4; ++r) {
        compute_dists(shX, shY, shRd, shRd + 9, shDist, shP, tid);
        float ca[1] = {0.0f};
        for (int i = tid; i < NSAMP; i += 256) {
          float c2 = (shDist[i] < INLIER_TH) ? 1.0f : 0.0f;
          shKey[i] = c2;   // candidate inlier mask (shKey reused)
          ca[0] += c2;
        }
        waveBlockReduce<1>(ca, sred, tid);
        if (tid == 0) sred[100] = ca[0];
        __syncthreads();
        float cnt = sred[100];
        float prev = sred[101];
        bool go = (cnt >= (float)NUM_CORR_) && (cnt > prev);
        __syncthreads();
        if (tid == 0) sred[101] = cnt;
        if (go)
          for (int i = tid; i < NSAMP; i += 256) shInl[i] = shKey[i];
        __syncthreads();
        procrustes_block(shX, shY, shInl, sred, shMu, shRn, tid);
        if (tid == 0 && go)
          for (int j = 0; j < 12; ++j) shRd[j] = shRn[j];
        __syncthreads();
      }

      // final Procrustes on refined inlier set + pose loss
      procrustes_block(shX, shY, shInl, sred, shMu, shRn, tid);
      if (tid == 0) {
        float dot = 0.0f;
        for (int i = 0; i < 3; ++i)
          for (int j = 0; j < 3; ++j)
            dot += shRn[i * 3 + j] * T[b * 16 + i * 4 + j];
        float cosang = fminf(fmaxf((dot - 1.0f) * 0.5f, -0.999999f), 0.999999f);
        float ang = acosf(cosang);
        float dt0 = shRn[9]  - T[b * 16 + 0 * 4 + 3];
        float dt1 = shRn[10] - T[b * 16 + 1 * 4 + 3];
        float dt2 = shRn[11] - T[b * 16 + 2 * 4 + 3];
        float terr = sqrtf(dt0 * dt0 + dt1 * dt1 + dt2 * dt2);
        float lr = SOFTCLIP_M * tanhf(ang / SOFTCLIP_M);
        float lt = SOFTCLIP_M * tanhf(terr / SOFTCLIP_M);
        shLv[kk] = 0.5f * (lr + lt);
      }
      __syncthreads();
    }

    // softmax over 8 hypotheses + null entry (tid 0)
    if (tid == 0) {
      float sc[IT_RANSAC_ + 1], lv[IT_RANSAC_ + 1];
      for (int k = 0; k < IT_RANSAC_; ++k) { sc[k] = shScores[k]; lv[k] = shLv[k]; }
      sc[IT_RANSAC_] = TH_OUT_ * (float)NSAMP;
      lv[IT_RANSAC_] = MAXLOSSNULL;
      float mx = sc[0];
      for (int k = 1; k <= IT_RANSAC_; ++k) mx = fmaxf(mx, sc[k]);
      float se = 0.0f, num = 0.0f;
      for (int k = 0; k <= IT_RANSAC_; ++k) {
        float e = __expf((sc[k] - mx) / SCORE_TEMP_);
        se += e;
        num += lv[k] * e;
      }
      total += num / se;
    }
    __syncthreads();
  }
  if (tid == 0) out[b] = total * (1.0f / (float)IT_MATCHES_);
}

// ============================================================================
extern "C" void kernel_launch(void* const* d_in, const int* in_sizes, int n_in,
                              void* d_out, int out_size, void* d_ws,
                              size_t ws_size, hipStream_t stream) {
  (void)in_sizes; (void)n_in; (void)out_size; (void)ws_size;
  const float* matches = (const float*)d_in[0];
  const float* kps0    = (const float*)d_in[1];
  const float* depth0  = (const float*)d_in[2];
  const float* kps1    = (const float*)d_in[3];
  const float* depth1  = (const float*)d_in[4];
  const float* K0      = (const float*)d_in[5];
  const float* K1      = (const float*)d_in[6];
  // d_in[7] = Kori_color0 (unused by reference forward)
  const float* T       = (const float*)d_in[8];

  const int total = IT_MATCHES_ * B_ * NSAMP;  // 65536
  char* ws = (char*)d_ws;
  int*   sidx_ws = (int*)ws;                                   // 256 KB
  float* w_ws    = (float*)(ws + (size_t)total * 4);           // 256 KB
  float* Xws     = (float*)(ws + (size_t)total * 8);           // 768 KB
  float* Yws     = (float*)(ws + (size_t)total * 8 + (size_t)total * 12);

  sample_kernel<<<(B_ * NSAMP) / 8, 256, 0, stream>>>(matches, sidx_ws, w_ws);
  backproject_kernel<<<total / 256, 256, 0, stream>>>(
      sidx_ws, kps0, depth0, kps1, depth1, K0, K1, Xws, Yws);
  ransac_kernel<<<B_, 256, 0, stream>>>(Xws, Yws, w_ws, T, (float*)d_out);
}
